// MultiHeadAttention_39273180954852
// MI455X (gfx1250) — matrix operand order repaired and checked by reading, not compiled, not run on hardware
//
#include <hip/hip_runtime.h>
#include <hip/hip_bf16.h>

// ---------------------------------------------------------------------------
// Event-gated multi-head attention for MI455X (gfx1250).
// wave32 + v_wmma_f32_16x16x32_f16 + TDM (tensor_load_to_lds) A-tile staging.
// Shapes: B=8, L=1024, D=512, H=8, DK=DV=64.
// ---------------------------------------------------------------------------

typedef _Float16 h16;
typedef __attribute__((ext_vector_type(16))) _Float16 v16h;
typedef __attribute__((ext_vector_type(8)))  _Float16 v8h;
typedef __attribute__((ext_vector_type(8)))  float    v8f;
typedef unsigned int u32x4 __attribute__((ext_vector_type(4)));
typedef int          i32x4 __attribute__((ext_vector_type(4)));
typedef int          i32x8 __attribute__((ext_vector_type(8)));

#define NB   8
#define NL   1024
#define ND   512
#define NH   8
#define NEGV (-1e9f)

#if __has_builtin(__builtin_amdgcn_tensor_load_to_lds)
#define HAVE_TDM 1
#else
#define HAVE_TDM 0
#endif

// Assemble a 16-bit WMMA fragment slice: 16 contiguous halves taken as two
// groups of 8 (p[0..7] and p[8..15]).
__device__ __forceinline__ v16h frag2x8(const h16* __restrict__ p) {
  v8h lo = *(const v8h*)p;
  v8h hi = *(const v8h*)(p + 8);
  v16h a;
#pragma unroll
  for (int i = 0; i < 8; ++i) { a[i] = lo[i]; a[8 + i] = hi[i]; }
  return a;
}

__device__ __forceinline__ v8f wmma16(v16h a, v16h b, v8f c) {
  return __builtin_amdgcn_wmma_f32_16x16x32_f16(false, a, false, b,
                                                (short)0, c, false, false);
}

#if HAVE_TDM
// TDM: DMA a [16 rows x 512 cols] f16 tile (row stride 512 elems) from global
// into LDS. 2-D descriptor: groups 2/3 zero (<=2D tensor).
__device__ __forceinline__ void tdm_load_a_tile(const h16* gsrc,
                                                void* lds_dst) {
  unsigned long long ga = (unsigned long long)(size_t)gsrc;
  unsigned lds = (unsigned)(size_t)lds_dst;  // low 32 bits = LDS byte offset
  u32x4 g0;
  g0.x = 1u;                                        // count=1, user mode
  g0.y = lds;                                       // lds_addr
  g0.z = (unsigned)ga;                              // global_addr[31:0]
  g0.w = (unsigned)((ga >> 32) & 0x1ffffffull) |    // global_addr[56:32]
         (2u << 30);                                // type=2 ("image")
  i32x8 g1;
  g1[0] = 0x10000;                 // workgroup_mask=0, data_size=1 (2 bytes)
  g1[1] = (int)(512u << 16);       // tensor_dim0[15:0]=512  (bits 63:48)
  g1[2] = (int)(8192u << 16);      // td0[31:16]=0, tensor_dim1[15:0]=8192
  g1[3] = (int)(512u << 16);       // td1[31:16]=0, tile_dim0=512
  g1[4] = 16;                      // tile_dim1=16, tile_dim2=0
  g1[5] = 512;                     // tensor_dim0_stride[31:0]=512
  g1[6] = 0;                       // stride0 hi, stride1 lo
  g1[7] = 0;                       // stride1 hi
  i32x4 gz = {0, 0, 0, 0};
#if defined(__clang_major__) && (__clang_major__ >= 23)
  i32x8 gz8 = {0, 0, 0, 0, 0, 0, 0, 0};
  __builtin_amdgcn_tensor_load_to_lds(g0, g1, gz, gz, gz8, 0);
#else
  __builtin_amdgcn_tensor_load_to_lds(g0, g1, gz, gz, 0);
#endif
}

__device__ __forceinline__ void tdm_wait() {
#if __has_builtin(__builtin_amdgcn_s_wait_tensorcnt)
  __builtin_amdgcn_s_wait_tensorcnt(0);
#else
  asm volatile("s_wait_tensorcnt 0x0" ::: "memory");
#endif
}
#endif  // HAVE_TDM

// ---------------------------------------------------------------------------
// 1) LayerNorm(q) -> f16   (one wave per 512-wide row, 8 rows per block)
// ---------------------------------------------------------------------------
__global__ __launch_bounds__(256) void ln_f16_kernel(
    const float* __restrict__ q, const float* __restrict__ gamma,
    const float* __restrict__ beta, h16* __restrict__ qn) {
  int row  = blockIdx.x * 8 + (threadIdx.x >> 5);
  int lane = threadIdx.x & 31;
  const float* x = q + (size_t)row * ND;
  float vals[16];
  float s = 0.f, s2 = 0.f;
#pragma unroll
  for (int i = 0; i < 16; ++i) {
    float t = x[lane + i * 32];
    vals[i] = t; s += t; s2 += t * t;
  }
#pragma unroll
  for (int m = 16; m >= 1; m >>= 1) {
    s  += __shfl_xor(s,  m, 32);
    s2 += __shfl_xor(s2, m, 32);
  }
  float mu  = s * (1.f / ND);
  float var = s2 * (1.f / ND) - mu * mu;
  float inv = rsqrtf(var + 1e-6f);
  h16* y = qn + (size_t)row * ND;
#pragma unroll
  for (int i = 0; i < 16; ++i) {
    int c = lane + i * 32;
    y[c] = (h16)((vals[i] - mu) * inv * gamma[c] + beta[c]);
  }
}

// ---------------------------------------------------------------------------
// 2) f32 -> f16 elementwise convert (k, v)
// ---------------------------------------------------------------------------
__global__ __launch_bounds__(256) void cvt_f16_kernel(
    const float* __restrict__ in, h16* __restrict__ out, int n) {
  int i = blockIdx.x * blockDim.x + threadIdx.x;
  if (i < n) out[i] = (h16)in[i];
}

// ---------------------------------------------------------------------------
// 3) Pack a [512 out x 512 in] f32 torch-Linear weight into WMMA-B fragment
//    layout of B = W^T (K=in, N=out):
//      P[((kt*32 + nt)*32 + lane)*16 + i] = W[n, k]
//      n = nt*16 + (lane&15),  k = kt*32 + (lane<16 ? 0 : 16) + i
// ---------------------------------------------------------------------------
__global__ __launch_bounds__(256) void pack_w_kernel(
    const float* __restrict__ W, h16* __restrict__ P) {
  int idx  = blockIdx.x * blockDim.x + threadIdx.x;  // 512*512 elements
  int i    = idx & 15;
  int lane = (idx >> 4) & 31;
  int nt   = (idx >> 9) & 31;
  int kt   = idx >> 14;
  int n = nt * 16 + (lane & 15);
  int k = kt * 32 + ((lane < 16) ? 0 : 16) + i;
  P[idx] = (h16)W[(size_t)n * ND + k];
}

// ---------------------------------------------------------------------------
// 4) GEMM: C[8192 x 512] = A[8192 x 512](f16) x Bpacked(f16), f32 accum.
//    One block = one 16-row tile, 8 waves = 8 x 64-wide N slices.
//    A tile (16 KB) staged once into LDS via TDM; B ping-pong from L2,
//    K loop fully unrolled so buffer indices are static (no v_mov rotation).
//    mode 0: store f16 to [B,H,L,64]            (q_en, q_ex, kh)
//    mode 1: scatter f16 to PV fragment layout  (vh)
//    mode 2: store f32 + bias + residual        (final projection)
// ---------------------------------------------------------------------------
__global__ __launch_bounds__(256) void gemm_kernel(
    const h16* __restrict__ A, const h16* __restrict__ Bp,
    void* __restrict__ dst, const float* __restrict__ bias,
    const float* __restrict__ residual, float alpha, int mode) {
  __shared__ h16 sA[16 * ND];  // 16 KiB
  int wv = threadIdx.x >> 5, lane = threadIdx.x & 31;
  int m0 = blockIdx.x * 16;    // row tile
  int n0 = wv * 64;            // per-wave N slice
  int kbA = (lane < 16) ? 0 : 16;
  int rl  = lane & 15;

  // ---- Stage the shared A tile into LDS ----
#if HAVE_TDM
  if (threadIdx.x < 32) {
    tdm_load_a_tile(A + (size_t)m0 * ND, (void*)&sA[0]);
    tdm_wait();
  }
#else
  {
    const v8h* src = (const v8h*)(A + (size_t)m0 * ND);
    v8h* dstl = (v8h*)&sA[0];
    for (int t = threadIdx.x; t < (16 * ND) / 8; t += 256) dstl[t] = src[t];
  }
#endif
  __syncthreads();

  // ---- K loop: fully unrolled ping-pong software pipeline ----
  v8f acc[4] = {};
  v16h bbuf[2][4];
  v16h abuf[2];
#pragma unroll
  for (int j = 0; j < 4; ++j)
    bbuf[0][j] =
        *(const v16h*)(Bp + (((size_t)((n0 >> 4) + j) * 32 + lane) << 4));
  abuf[0] = frag2x8(&sA[rl * ND + kbA]);

#pragma unroll
  for (int kt = 0; kt < 16; ++kt) {
    const int cur = kt & 1, nx = cur ^ 1;
    if (kt + 1 < 16) {
#pragma unroll
      for (int j = 0; j < 4; ++j)
        bbuf[nx][j] = *(const v16h*)(
            Bp + (((size_t)((kt + 1) * 32 + (n0 >> 4) + j) * 32 + lane) << 4));
      abuf[nx] = frag2x8(&sA[rl * ND + (kt + 1) * 32 + kbA]);
    }
#pragma unroll
    for (int j = 0; j < 4; ++j) acc[j] = wmma16(abuf[cur], bbuf[cur][j], acc[j]);
  }

  int mb = (lane < 16) ? 0 : 8;
#pragma unroll
  for (int j = 0; j < 4; ++j) {
#pragma unroll
    for (int r = 0; r < 8; ++r) {
      float val = acc[j][r] * alpha;
      int m = m0 + mb + r;
      int n = n0 + j * 16 + rl;
      if (mode == 0) {
        int b = m >> 10, l = m & 1023, h = n >> 6, dk = n & 63;
        ((h16*)dst)[(((size_t)(b * NH + h) * NL + l) << 6) + dk] = (h16)val;
      } else if (mode == 1) {
        int b = m >> 10, kk = m & 1023, h = n >> 6, d = n & 63;
        int kt2 = kk >> 5, rem = kk & 31, half = rem >> 4, i2 = rem & 15;
        int nt2 = d >> 4, lane2 = (d & 15) + (half << 4);
        size_t bh = (size_t)b * NH + h;
        ((h16*)dst)[((((bh * 32 + kt2) * 4 + nt2) * 32 + lane2) << 4) + i2] =
            (h16)val;
      } else {
        ((float*)dst)[(size_t)m * ND + n] =
            val + bias[n] + residual[(size_t)m * ND + n];
      }
    }
  }
}

// ---------------------------------------------------------------------------
// 5) Attention: one block per (b, h, 16-row q tile). 8 waves.
//    Phase A: scores via WMMA (per-lane q_en/q_ex row select) -> LDS (f32)
//    Phase B: softmax (16 threads per row), write attn f32 + probs f16
//    Phase C: PV via WMMA (waves 0..3), store heads [B, L, H*64] f16
// ---------------------------------------------------------------------------
__global__ __launch_bounds__(256) void attn_kernel(
    const h16* __restrict__ q_en, const h16* __restrict__ q_ex,
    const h16* __restrict__ kh, const h16* __restrict__ vp,
    const int* __restrict__ etype, const unsigned char* __restrict__ msk,
    float* __restrict__ attn_out, h16* __restrict__ heads) {
  extern __shared__ char smem[];
  float* sc = (float*)smem;                  // 16*1024*4 = 64 KiB
  h16*   pr = (h16*)(smem + 16 * NL * 4);    // 16*1024*2 = 32 KiB

  int qt = blockIdx.x & 63;
  int bh = blockIdx.x >> 6;
  int b  = bh >> 3;
  int h  = bh & 7;
  int q0 = qt << 4;
  int wv = threadIdx.x >> 5, lane = threadIdx.x & 31;
  int rl  = lane & 15;
  int kbA = (lane < 16) ? 0 : 16;
  int kbB = (lane < 16) ? 0 : 16;
  int mb  = (lane < 16) ? 0 : 8;

  // Event flags for the 8 C-rows owned by this lane (bitmask over r).
  unsigned ren_mask = 0;
#pragma unroll
  for (int r = 0; r < 8; ++r)
    ren_mask |= (etype[b * NL + q0 + mb + r] != 0) ? (1u << r) : 0u;

  // ---- A fragments: per-lane row selects en vs ex projection ----
  int row_is_en = (etype[b * NL + q0 + rl] != 0);
  const h16* abase = (row_is_en ? q_en : q_ex) +
                     (((size_t)bh * NL + q0 + rl) << 6);
  v16h a0 = frag2x8(abase + kbA);
  v16h a1 = frag2x8(abase + 32 + kbA);

  // ---- Phase A: scores, fully unrolled ping-pong over 8 column tiles ----
  const h16* kcol = kh + (((size_t)bh * NL + wv * 128 + rl) << 6) + kbB;
  v16h bb[2][2];
  bb[0][0] = *(const v16h*)(kcol);
  bb[0][1] = *(const v16h*)(kcol + 32);
#pragma unroll
  for (int j = 0; j < 8; ++j) {
    const int cur = j & 1, nx = cur ^ 1;
    if (j + 1 < 8) {
      const h16* knxt = kcol + ((size_t)(j + 1) << 10);  // +16 rows * 64
      bb[nx][0] = *(const v16h*)(knxt);
      bb[nx][1] = *(const v16h*)(knxt + 32);
    }
    v8f c = {};
    c = wmma16(a0, bb[cur][0], c);
    c = wmma16(a1, bb[cur][1], c);

    int n0 = wv * 128 + j * 16;
    int ki = n0 + rl;
    int col_en = (etype[b * NL + ki] != 0);
#pragma unroll
    for (int r = 0; r < 8; ++r) {
      int qrow = mb + r;
      unsigned char mby = msk[(((size_t)(b * NL + q0 + qrow)) << 10) + ki];
      int ren = (ren_mask >> r) & 1;
      float s = !col_en ? 0.f : ((ren && mby) ? NEGV : c[r]);
      sc[qrow * NL + ki] = s;
    }
  }
  __syncthreads();

  // ---- Phase B: softmax over 1024 cols; 16 threads per row ----
  {
    int row = threadIdx.x >> 4, sub = threadIdx.x & 15;
    float* srow = sc + row * NL;
    float mx = -3.4e38f;
    for (int i = 0; i < 64; ++i) mx = fmaxf(mx, srow[sub + i * 16]);
#pragma unroll
    for (int m = 8; m >= 1; m >>= 1) mx = fmaxf(mx, __shfl_xor(mx, m, 32));
    float sm = 0.f;
    for (int i = 0; i < 64; ++i) sm += __expf(srow[sub + i * 16] - mx);
#pragma unroll
    for (int m = 8; m >= 1; m >>= 1) sm += __shfl_xor(sm, m, 32);
    float inv = 1.f / sm;
    float* arow = attn_out + (((size_t)bh * NL + q0 + row) << 10);
    h16* prow = pr + row * NL;
    for (int i = 0; i < 64; ++i) {
      int c = sub + i * 16;
      float p = __expf(srow[c] - mx) * inv;
      arow[c] = p;
      prow[c] = (h16)p;
    }
  }
  __syncthreads();

  // ---- Phase C: PV (waves 0..3, one 16-wide dv tile each), ping-pong ----
  if (wv < 4) {
    v8f c = {};
    const h16* prow = pr + rl * NL;
    const h16* vbase = vp + ((((size_t)bh * 32) * 4 + wv) * 32 + lane) * 16;
    v16h vb[2];
    vb[0] = *(const v16h*)vbase;
#pragma unroll 2
    for (int kt = 0; kt < 32; ++kt) {
      const int cur = kt & 1, nx = cur ^ 1;
      if (kt + 1 < 32)
        vb[nx] = *(const v16h*)(vbase + (size_t)(kt + 1) * 2048);
      v16h a = frag2x8(prow + kt * 32 + kbA);
      c = wmma16(a, vb[cur], c);
    }
#pragma unroll
    for (int r = 0; r < 8; ++r) {
      int qrow = mb + r;
      heads[(((size_t)(b * NL + q0 + qrow)) << 9) + (h << 6) + (wv << 4) + rl] =
          (h16)c[r];
    }
  }
}

// ---------------------------------------------------------------------------
// Host-side launch
// ---------------------------------------------------------------------------
extern "C" void kernel_launch(void* const* d_in, const int* in_sizes, int n_in,
                              void* d_out, int out_size, void* d_ws,
                              size_t ws_size, hipStream_t stream) {
  const float* q     = (const float*)d_in[0];
  const float* k     = (const float*)d_in[1];
  const float* v     = (const float*)d_in[2];
  const int*   etype = (const int*)d_in[3];
  const unsigned char* msk = (const unsigned char*)d_in[4];
  const float* w_qs  = (const float*)d_in[5];
  const float* w_ex  = (const float*)d_in[6];
  const float* w_ks  = (const float*)d_in[7];
  const float* w_vs  = (const float*)d_in[8];
  const float* w_fc  = (const float*)d_in[9];
  const float* b_fc  = (const float*)d_in[10];
  const float* gamma = (const float*)d_in[11];
  const float* beta  = (const float*)d_in[12];

  const size_t TOK = (size_t)NB * NL;        // 8192
  const size_t MAT = TOK * ND;               // 4,194,304 elems
  const size_t MAT2 = MAT * sizeof(h16);     // 8 MiB
  const size_t WP = (size_t)ND * ND * sizeof(h16);  // 512 KiB

  char* ws = (char*)d_ws;
  h16* qn_h   = (h16*)ws; ws += MAT2;
  h16* k_h    = (h16*)ws; ws += MAT2;
  h16* v_h    = (h16*)ws; ws += MAT2;
  h16* qen_h  = (h16*)ws; ws += MAT2;   // [B,H,L,64]
  h16* qex_h  = (h16*)ws; ws += MAT2;   // [B,H,L,64]
  h16* kh_h   = (h16*)ws; ws += MAT2;   // [B,H,L,64]
  h16* vh_p   = (h16*)ws; ws += MAT2;   // PV fragment layout
  h16* hd_h   = (h16*)ws; ws += MAT2;   // heads [B,L,512]
  h16* wq_p   = (h16*)ws; ws += WP;
  h16* wqx_p  = (h16*)ws; ws += WP;
  h16* wk_p   = (h16*)ws; ws += WP;
  h16* wv_p   = (h16*)ws; ws += WP;
  h16* wfc_p  = (h16*)ws; ws += WP;

  float* out_f  = (float*)d_out;             // [B,L,D]
  float* attn_f = out_f + MAT;               // [B,H,L,L]

  // 1) LayerNorm + converts
  ln_f16_kernel<<<TOK / 8, 256, 0, stream>>>(q, gamma, beta, qn_h);
  cvt_f16_kernel<<<(MAT + 255) / 256, 256, 0, stream>>>(k, k_h, (int)MAT);
  cvt_f16_kernel<<<(MAT + 255) / 256, 256, 0, stream>>>(v, v_h, (int)MAT);

  // 2) Pack weights into WMMA-B fragment layout
  const int PGRID = (ND * ND) / 256;
  pack_w_kernel<<<PGRID, 256, 0, stream>>>(w_qs, wq_p);
  pack_w_kernel<<<PGRID, 256, 0, stream>>>(w_ex, wqx_p);
  pack_w_kernel<<<PGRID, 256, 0, stream>>>(w_ks, wk_p);
  pack_w_kernel<<<PGRID, 256, 0, stream>>>(w_vs, wv_p);
  pack_w_kernel<<<PGRID, 256, 0, stream>>>(w_fc, wfc_p);

  // 3) Projections (1/sqrt(dk) folded into q projections)
  const int GGRID = (int)(TOK / 16);  // 512 blocks: 1 row tile each, 8 waves
  gemm_kernel<<<GGRID, 256, 0, stream>>>(qn_h, wq_p,  (void*)qen_h, nullptr, nullptr, 0.125f, 0);
  gemm_kernel<<<GGRID, 256, 0, stream>>>(qn_h, wqx_p, (void*)qex_h, nullptr, nullptr, 0.125f, 0);
  gemm_kernel<<<GGRID, 256, 0, stream>>>(k_h,  wk_p,  (void*)kh_h,  nullptr, nullptr, 1.0f,   0);
  gemm_kernel<<<GGRID, 256, 0, stream>>>(v_h,  wv_p,  (void*)vh_p,  nullptr, nullptr, 1.0f,   1);

  // 4) Attention (scores + mask/blend + softmax + PV), 96 KiB dynamic LDS
  const int AGRID = NB * NH * (NL / 16);  // 4096
  const size_t LDS = 16 * NL * 4 + 16 * NL * 2;  // 98304 B
  attn_kernel<<<AGRID, 256, LDS, stream>>>(qen_h, qex_h, kh_h, vh_p, etype,
                                           msk, attn_f, hd_h);

  // 5) Output projection + bias + residual
  gemm_kernel<<<GGRID, 256, 0, stream>>>(hd_h, wfc_p, (void*)out_f, b_fc, q,
                                         1.0f, 2);
}
